// CascadeMVSNet_10350871184004
// MI455X (gfx1250) — matrix operand order repaired and checked by reading, not compiled, not run on hardware
//
#include <hip/hip_runtime.h>
#include <hip/hip_bf16.h>

// ---------------- problem dims ----------------
#define BB 2
#define VS 2          // V_SRC
#define CC 32
#define DD 48
#define HH 128
#define WW 160
#define NVIEWS 3      // V_SRC + 1

typedef __attribute__((ext_vector_type(16))) _Float16 v16h;
typedef __attribute__((ext_vector_type(8)))  _Float16 v8h;
typedef __attribute__((ext_vector_type(4)))  _Float16 v4h;
typedef __attribute__((ext_vector_type(8)))  float    v8f;
typedef __attribute__((ext_vector_type(4)))  float    v4f;

// ---------------------------------------------------------------
// 1) projection composition:  rot/trans of  src_P @ inv(ref_P)
// ---------------------------------------------------------------
__device__ __forceinline__ void mat3_inv(const float* M, float* o) {
    float a=M[0],b=M[1],c=M[2],d=M[3],e=M[4],f=M[5],g=M[6],h=M[7],i=M[8];
    float A =  e*i - f*h;
    float Bc = -(d*i - f*g);
    float Cc =  d*h - e*g;
    float det = a*A + b*Bc + c*Cc;
    float id = 1.0f/det;
    o[0]=A*id;  o[1]=(c*h-b*i)*id; o[2]=(b*f-c*e)*id;
    o[3]=Bc*id; o[4]=(a*i-c*g)*id; o[5]=(c*d-a*f)*id;
    o[6]=Cc*id; o[7]=(b*g-a*h)*id; o[8]=(a*e-b*d)*id;
}

// compose: M = K[:3,:3] @ E[:3,:3],  m = K[:3,:3] @ E[:3,3]
__device__ __forceinline__ void compose(const float* E, const float* K,
                                        float* M, float* m) {
    for (int i = 0; i < 3; ++i) {
        for (int j = 0; j < 3; ++j) {
            float s = 0.f;
            for (int k = 0; k < 3; ++k) s += K[i*4+k]*E[k*4+j];
            M[i*3+j] = s;
        }
        float s = 0.f;
        for (int k = 0; k < 3; ++k) s += K[i*4+k]*E[k*4+3];
        m[i] = s;
    }
}

__global__ void proj_kernel(const float* __restrict__ ref_proj,
                            const float* __restrict__ src_projs,
                            float* __restrict__ rt /* (B*VS)*12 */) {
    int t = threadIdx.x;
    if (t >= BB*VS) return;
    int b = t / VS, v = t % VS;

    const float* rp = ref_proj + (size_t)b*32;          // E = rp, K = rp+16
    float Mr[9], mr[3];
    compose(rp, rp+16, Mr, mr);

    const float* sp = src_projs + ((size_t)(b*VS+v))*32;
    float Ms[9], ms[3];
    compose(sp, sp+16, Ms, ms);

    float Mi[9];
    mat3_inv(Mr, Mi);
    float ti[3];
    for (int i = 0; i < 3; ++i) {
        float s = 0.f;
        for (int j = 0; j < 3; ++j) s += Mi[i*3+j]*mr[j];
        ti[i] = -s;
    }
    float* o = rt + t*12;
    for (int i = 0; i < 3; ++i) {
        for (int j = 0; j < 3; ++j) {
            float s = 0.f;
            for (int k = 0; k < 3; ++k) s += Ms[i*3+k]*Mi[k*3+j];
            o[i*3+j] = s;
        }
        float s = 0.f;
        for (int k = 0; k < 3; ++k) s += Ms[i*3+k]*ti[k];
        o[9+i] = s + ms[i];
    }
}

// ---------------------------------------------------------------
// 2) pack conv weights to fp16 B-fragment order: wts[t*32 + p]
//    (B-matrix 16-bit layout: lane group g, element e reads K = g*8 + e (e<8), 16 + g*8 + (e-8) (e>=8))
// ---------------------------------------------------------------
__global__ void weight_kernel(const float* __restrict__ w_reg,
                              _Float16* __restrict__ wts) {
    int i = blockIdx.x*blockDim.x + threadIdx.x;
    if (i >= 27*CC) return;
    int t = i / CC;        // tap index kd*9+kh*3+kw
    int k = (i & 7) + ((i >> 3) & 1)*16 + ((i >> 4) & 1)*8;        // channel
    wts[i] = (_Float16)w_reg[k*27 + t];
}

// ---------------------------------------------------------------
// 3) channel-last transposes
// ---------------------------------------------------------------
__global__ void cl_ref_kernel(const float* __restrict__ src,
                              float* __restrict__ dst) {
    size_t i = (size_t)blockIdx.x*blockDim.x + threadIdx.x;
    const size_t total = (size_t)BB*HH*WW*CC;
    if (i >= total) return;
    int c = i % CC;
    int w = (i / CC) % WW;
    int h = (i / ((size_t)CC*WW)) % HH;
    int b = i / ((size_t)CC*WW*HH);
    dst[i] = src[(((size_t)b*CC + c)*HH + h)*WW + w];
}

__global__ void cl_src_kernel(const float* __restrict__ src,
                              float* __restrict__ dst) {
    size_t i = (size_t)blockIdx.x*blockDim.x + threadIdx.x;
    const size_t total = (size_t)BB*VS*HH*WW*CC;
    if (i >= total) return;
    int c = i % CC;
    int w = (i / CC) % WW;
    int h = (i / ((size_t)CC*WW)) % HH;
    int v = (i / ((size_t)CC*WW*HH)) % VS;
    int b = i / ((size_t)CC*WW*HH*VS);
    dst[i] = src[((((size_t)b*VS + v)*CC + c)*HH + h)*WW + w];
}

// ---------------------------------------------------------------
// 4) homography warp + variance, fp16 channel-last output
// ---------------------------------------------------------------
__global__ void warp_var_kernel(const float* __restrict__ refCL,
                                const float* __restrict__ srcCL,
                                const float* __restrict__ rt,
                                const float* __restrict__ dvals,
                                _Float16* __restrict__ varCL) {
    size_t idx = (size_t)blockIdx.x*blockDim.x + threadIdx.x;
    const size_t total = (size_t)BB*DD*HH*WW;
    if (idx >= total) return;
    int w = idx % WW;
    int h = (idx / WW) % HH;
    int b = idx / ((size_t)WW*HH*DD);

    float depth = dvals[idx];

    int   vidx[VS][4];
    float vwt[VS][4];
#pragma unroll
    for (int v = 0; v < VS; ++v) {
        const float* p = rt + (b*VS + v)*12;
        float fx = (float)w, fy = (float)h;
        float rx = p[0]*fx + p[1]*fy + p[2];
        float ry = p[3]*fx + p[4]*fy + p[5];
        float rz = p[6]*fx + p[7]*fy + p[8];
        float X = rx*depth + p[9];
        float Y = ry*depth + p[10];
        float Z = rz*depth + p[11];
        float iz = 1.0f / Z;
        float px = X*iz, py = Y*iz;
        float x0 = floorf(px), y0 = floorf(py);
        int t = 0;
#pragma unroll
        for (int dx = 0; dx <= 1; ++dx)
#pragma unroll
            for (int dy = 0; dy <= 1; ++dy) {
                float xs = x0 + dx, ys = y0 + dy;
                float wg = (1.0f - fabsf(px - xs))*(1.0f - fabsf(py - ys));
                bool valid = (xs >= 0.f) && (xs <= (float)(WW-1)) &&
                             (ys >= 0.f) && (ys <= (float)(HH-1));
                int xi = (int)fminf(fmaxf(xs, 0.f), (float)(WW-1));
                int yi = (int)fminf(fmaxf(ys, 0.f), (float)(HH-1));
                vidx[v][t] = yi*WW + xi;
                vwt[v][t]  = valid ? wg : 0.f;
                ++t;
            }
    }

    const float* rp = refCL + ((size_t)(b*HH + h)*WW + w)*CC;
    _Float16* op = varCL + idx*CC;
    const float inv3 = 1.0f/3.0f;

#pragma unroll
    for (int c = 0; c < CC; c += 4) {
        v4f r = *(const v4f*)(rp + c);
        v4f s = r;
        v4f q = r*r;
#pragma unroll
        for (int v = 0; v < VS; ++v) {
            const float* sb = srcCL + ((size_t)(b*VS + v)*HH*WW)*CC + c;
            v4f wv = {0.f, 0.f, 0.f, 0.f};
#pragma unroll
            for (int t = 0; t < 4; ++t) {
                v4f sv = *(const v4f*)(sb + (size_t)vidx[v][t]*CC);
                float wt = vwt[v][t];
                wv += sv*wt;
            }
            s += wv;
            q += wv*wv;
        }
        v4f mn = s*inv3;
        v4f var = q*inv3 - mn*mn;
        v4h hv = {(_Float16)var.x, (_Float16)var.y,
                  (_Float16)var.z, (_Float16)var.w};
        *(v4h*)(op + c) = hv;
    }
}

// ---------------------------------------------------------------
// 5) 3-D conv via WMMA with LDS-staged halo tile.
//    Workgroup (4 waves) computes an 8(h) x 16(w) patch at fixed (b,d).
//    Halo: 3(d) x 10(h) x 18(w) voxels x 32 fp16 channels staged once,
//    padded to 40 halves/voxel (80B stride -> conflict-free ds_load_b128,
//    16B-aligned fragments). Each wave: 2 rows x 27 taps of
//    v_wmma_f32_16x16x32_f16 fed entirely from LDS.
// ---------------------------------------------------------------
#define LD3 3
#define LH  10          // 8 rows + halo
#define LW  18          // 16 cols + halo
#define VSTR 40         // padded halves per voxel (80 B)
#define NVOX (LD3*LH*LW) // 540

__global__ void __launch_bounds__(128)
conv_wmma_kernel(const _Float16* __restrict__ varCL,
                 const _Float16* __restrict__ wts,
                 const float* __restrict__ breg,
                 float* __restrict__ cost) {
    __shared__ _Float16 sVar[NVOX * VSTR];   // 43,200 B
    __shared__ _Float16 sW[27 * CC];         //  1,728 B

    const int WT = WW/16;                    // 10 tiles per row
    const int HB = HH/8;                     // 16 h-blocks

    int tid  = threadIdx.x;
    int lane = tid & 31;
    int wv   = tid >> 5;                     // wave id 0..3

    int bw   = blockIdx.x % WT;
    int hblk = (blockIdx.x / WT) % HB;
    int d    = (blockIdx.x / (WT*HB)) % DD;
    int b    =  blockIdx.x / (WT*HB*DD);
    int w0   = bw*16;
    int h0   = hblk*8;

    // ---- stage halo tile (zero-padded) ----
    for (int i = tid; i < NVOX; i += 128) {
        int wxl = i % LW;
        int hyl = (i / LW) % LH;
        int dzl = i / (LW*LH);
        int dz = d + dzl - 1;
        int hy = h0 + hyl - 1;
        int wx = w0 + wxl - 1;
        bool ok = ((unsigned)dz < (unsigned)DD) &&
                  ((unsigned)hy < (unsigned)HH) &&
                  ((unsigned)wx < (unsigned)WW);
        v16h x0 = {};
        v16h x1 = {};
        if (ok) {
            const _Float16* gp = varCL +
                ((((size_t)b*DD + dz)*HH + hy)*WW + wx)*CC;
            x0 = *(const v16h*)gp;
            x1 = *(const v16h*)(gp + 16);
        }
        *(v16h*)(sVar + (size_t)i*VSTR)      = x0;
        *(v16h*)(sVar + (size_t)i*VSTR + 16) = x1;
    }
    for (int i = tid; i < 27*CC; i += 128) sW[i] = wts[i];
    __syncthreads();

    int m = lane & 15;       // output voxel within row (M)
    int g = lane >> 4;       // lane group

    float bias = breg[0];

#pragma unroll
    for (int r = 0; r < 2; ++r) {
        int hl = wv*2 + r;                   // output row (local)
        v8f acc = {};
#pragma unroll
        for (int kd = 0; kd < 3; ++kd)
#pragma unroll
            for (int kh = 0; kh < 3; ++kh)
#pragma unroll
                for (int kw = 0; kw < 3; ++kw) {
                    int lvox = (kd*LH + (hl + kh))*LW + (m + kw);
                    const _Float16* lp = sVar + (size_t)lvox*VSTR + g*8;
                    v8h lo = *(const v8h*)lp;        // K = g*8 + {0..7}
                    v8h hi = *(const v8h*)(lp + 16); // K = g*8 + {16..23}
                    v16h a;
#pragma unroll
                    for (int e = 0; e < 8; ++e) { a[e] = lo[e]; a[8+e] = hi[e]; }

                    int t = (kd*3 + kh)*3 + kw;
                    v16h bm = *(const v16h*)(sW + t*32 + g*16); // K = g*8+e (e<8), 16+g*8+(e-8) (e>=8)
                    acc = __builtin_amdgcn_wmma_f32_16x16x32_f16(
                            false, a, false, bm, (short)0, acc, false, false);
                }

        if (m == 0) {  // lane 0: M=0..7 at N=0; lane 16: M=8..15
            float* cp = cost + (((size_t)b*DD + d)*HH + (h0 + hl))*WW + w0 + g*8;
            v4f lo4 = {acc[0]+bias, acc[1]+bias, acc[2]+bias, acc[3]+bias};
            v4f hi4 = {acc[4]+bias, acc[5]+bias, acc[6]+bias, acc[7]+bias};
            *(v4f*)cp       = lo4;
            *(v4f*)(cp + 4) = hi4;
        }
    }
}

// ---------------------------------------------------------------
// 6) softmax over depth, expected depth, confidence
// ---------------------------------------------------------------
__global__ void softmax_kernel(const float* __restrict__ cost,
                               const float* __restrict__ dvals,
                               float* __restrict__ outDepth,
                               float* __restrict__ outConf,
                               float* __restrict__ outProb) {
    size_t idx = (size_t)blockIdx.x*blockDim.x + threadIdx.x;
    const size_t total = (size_t)BB*HH*WW;
    if (idx >= total) return;
    int b = idx / ((size_t)HH*WW);
    int pix = idx % ((size_t)HH*WW);

    const size_t plane = (size_t)HH*WW;
    const float* cp = cost + (size_t)b*DD*plane + pix;

    float p[DD];
    float mx = -3.4e38f;
#pragma unroll
    for (int d = 0; d < DD; ++d) {
        float v = cp[(size_t)d*plane];
        p[d] = v;
        mx = fmaxf(mx, v);
    }
    float s = 0.f;
#pragma unroll
    for (int d = 0; d < DD; ++d) {
        float e = __expf(p[d] - mx);
        p[d] = e;
        s += e;
    }
    float inv = 1.0f / s;

    const float* dvp = dvals + (size_t)b*DD*plane + pix;
    float dep = 0.f, di = 0.f;
#pragma unroll
    for (int d = 0; d < DD; ++d) {
        float pr = p[d]*inv;
        p[d] = pr;
        outProb[(size_t)b*DD*plane + (size_t)d*plane + pix] = pr;
        dep += pr * dvp[(size_t)d*plane];
        di  += pr * (float)d;
    }
    int dI = (int)di;
    dI = dI < 0 ? 0 : (dI > DD-1 ? DD-1 : dI);
    float conf = 0.f;
#pragma unroll
    for (int d = 0; d < DD; ++d)
        conf += ((d == dI) || (d == dI+1)) ? p[d] : 0.f;

    outDepth[idx] = dep;
    outConf[idx]  = conf;
}

// ---------------------------------------------------------------
// host-side launch
// ---------------------------------------------------------------
extern "C" void kernel_launch(void* const* d_in, const int* in_sizes, int n_in,
                              void* d_out, int out_size, void* d_ws, size_t ws_size,
                              hipStream_t stream) {
    const float* ref_feature  = (const float*)d_in[0];
    const float* src_features = (const float*)d_in[1];
    const float* ref_proj     = (const float*)d_in[2];
    const float* src_projs    = (const float*)d_in[3];
    const float* depth_values = (const float*)d_in[4];
    const float* w_reg        = (const float*)d_in[5];
    const float* b_reg        = (const float*)d_in[6];
    (void)in_sizes; (void)n_in; (void)out_size; (void)ws_size;

    char* ws = (char*)d_ws;
    const size_t OFF_RT   = 0;                                     // 48 f32
    const size_t OFF_W    = 1024;                                  // 864 f16
    const size_t OFF_REF  = 8192;
    const size_t SZ_REF   = (size_t)BB*HH*WW*CC*sizeof(float);     // 5.24 MB
    const size_t OFF_SRC  = OFF_REF + SZ_REF;
    const size_t SZ_SRC   = (size_t)BB*VS*HH*WW*CC*sizeof(float);  // 10.5 MB
    const size_t OFF_VAR  = OFF_SRC + SZ_SRC;
    const size_t SZ_VAR   = (size_t)BB*DD*HH*WW*CC*sizeof(_Float16); // 126 MB
    const size_t OFF_COST = OFF_VAR + SZ_VAR;                      // 7.9 MB

    float*    rt    = (float*)(ws + OFF_RT);
    _Float16* wts   = (_Float16*)(ws + OFF_W);
    float*    refCL = (float*)(ws + OFF_REF);
    float*    srcCL = (float*)(ws + OFF_SRC);
    _Float16* varCL = (_Float16*)(ws + OFF_VAR);
    float*    cost  = (float*)(ws + OFF_COST);

    float* outDepth = (float*)d_out;
    float* outConf  = outDepth + (size_t)BB*HH*WW;
    float* outProb  = outConf  + (size_t)BB*HH*WW;

    proj_kernel<<<1, 32, 0, stream>>>(ref_proj, src_projs, rt);

    {
        int n = 27*CC;
        weight_kernel<<<(n+255)/256, 256, 0, stream>>>(w_reg, wts);
    }
    {
        size_t n = (size_t)BB*HH*WW*CC;
        cl_ref_kernel<<<(unsigned)((n+255)/256), 256, 0, stream>>>(ref_feature, refCL);
    }
    {
        size_t n = (size_t)BB*VS*HH*WW*CC;
        cl_src_kernel<<<(unsigned)((n+255)/256), 256, 0, stream>>>(src_features, srcCL);
    }
    {
        size_t n = (size_t)BB*DD*HH*WW;
        warp_var_kernel<<<(unsigned)((n+255)/256), 256, 0, stream>>>(
            refCL, srcCL, rt, depth_values, varCL);
    }
    {
        int blocks = BB*DD*(HH/8)*(WW/16);   // 15,360 blocks of 128
        conv_wmma_kernel<<<blocks, 128, 0, stream>>>(varCL, wts, b_reg, cost);
    }
    {
        size_t n = (size_t)BB*HH*WW;
        softmax_kernel<<<(unsigned)((n+255)/256), 256, 0, stream>>>(
            cost, depth_values, outDepth, outConf, outProb);
    }
}
